// SpatialPatchMoE_55705725829897
// MI455X (gfx1250) — matrix operand order repaired and checked by reading, not compiled, not run on hardware
//
#include <hip/hip_runtime.h>

// ---------------------------------------------------------------------------
// SpatialPatchMoE for gfx1250 (MI455X): top-2 MoE of GatedConvBlocks.
// B=2, C=64, L=8, H=W=160, P=8 -> 800 patches, 20x20 per image.
// Pointwise layers run on v_wmma_f32_16x16x32_f16; conv/LN on VALU with
// fully-register-resident, fully-unrolled scatter convolution.
// ---------------------------------------------------------------------------

typedef __attribute__((ext_vector_type(4)))  float    v4f;
typedef __attribute__((ext_vector_type(8)))  float    v8f;
typedef __attribute__((ext_vector_type(8)))  _Float16 v8h;
typedef __attribute__((ext_vector_type(16))) _Float16 v16h;

#define NPATCH 800
#define NH 20
#define NW 20
#define CH 64
#define LL 8
#define HW 160
#define PLANE 25600            // HW*HW
#define YSTR 72                // padded fp16 row stride (multiple of 8 halves)
#define WSTR 136               // padded stride for WinT [64][128]
#define OSTR 72                // padded stride for WoutT [64][64]

struct Route { int e0, e1; float w0, w1; };

// ---------------------------- router ---------------------------------------
__global__ __launch_bounds__(256)
void moe_router(const float* __restrict__ x, const float* __restrict__ rw,
                const float* __restrict__ rb, Route* __restrict__ route)
{
  __shared__ float part[256];
  __shared__ float meanv[CH];
  __shared__ float lg[8];

  const int n   = blockIdx.x;
  const int b   = n / (NH * NW);
  const int rem = n % (NH * NW);
  const int ph  = rem / NW, pw = rem % NW;
  const int tid = threadIdx.x;
  const int c   = tid >> 2, q = tid & 3;

  const float* xin = x + ((long)(b * CH + c) * LL) * PLANE
                       + (long)(ph * 8) * HW + pw * 8;
  float s = 0.f;
#pragma unroll
  for (int ls = 0; ls < 2; ++ls) {
    const float* pl = xin + (long)(2 * q + ls) * PLANE;
#pragma unroll
    for (int py = 0; py < 8; ++py) {
      const v4f* r = (const v4f*)(pl + py * HW);   // 16B aligned: patch base % 8 floats == 0
      v4f r0 = r[0], r1 = r[1];
      s += r0[0] + r0[1] + r0[2] + r0[3] + r1[0] + r1[1] + r1[2] + r1[3];
    }
  }
  part[tid] = s;
  __syncthreads();
  if (q == 0)
    meanv[c] = (part[tid] + part[tid + 1] + part[tid + 2] + part[tid + 3]) * (1.f / 512.f);
  __syncthreads();
  if (tid < 8) {
    float acc = rb[tid];
    for (int cc = 0; cc < CH; ++cc) acc += meanv[cc] * rw[tid * CH + cc];
    lg[tid] = acc;
  }
  __syncthreads();
  if (tid == 0) {
    int e0 = 0; float v0 = lg[0];
    for (int ee = 1; ee < 8; ++ee) if (lg[ee] > v0) { v0 = lg[ee]; e0 = ee; }
    int e1 = (e0 == 0) ? 1 : 0; float v1 = lg[e1];
    for (int ee = 0; ee < 8; ++ee)
      if (ee != e0 && lg[ee] > v1) { v1 = lg[ee]; e1 = ee; }
    float x1  = __expf(v1 - v0);
    float inv = __builtin_amdgcn_rcpf(1.f + x1);
    Route r; r.e0 = e0; r.e1 = e1; r.w0 = inv; r.w1 = x1 * inv;
    route[n] = r;
  }
}

// ---------------------- WMMA fragment loaders -------------------------------
// A (16x32 f16): lanes 0-15 hold M=0-15 / lanes 16-31 same M, K offset +8;
// halves 0-7 -> K = kbase..kbase+7, halves 8-15 -> K = kbase+16..+23.
__device__ inline v16h load_fragA(const _Float16* base, int mt, int kt, int lane)
{
  const int m  = 16 * mt + (lane & 15);
  const int kb = 32 * kt + ((lane & 16) >> 1);
  const _Float16* p = base + m * YSTR + kb;
  v8h lo = *(const v8h*)p;
  v8h hi = *(const v8h*)(p + 16);
  v16h a;
#pragma unroll
  for (int i = 0; i < 8; ++i) { a[i] = lo[i]; a[i + 8] = hi[i]; }
  return a;
}

// B (32x16 f16): lane%16 = N column; halves 0-7 -> K = kbase..+7, 8-15 -> K = kbase+16..+23 (kbase = 32*kt + 8*(lane/16)).
__device__ inline v16h load_fragB(const _Float16* wT, int stride, int nt, int kt, int lane)
{
  const int n  = 16 * nt + (lane & 15);
  const int kb = 32 * kt + ((lane & 16) >> 1);
  v16h bf;
#pragma unroll
  for (int i = 0; i < 8; ++i) {
    bf[i]     = wT[(kb + i) * stride + n];
    bf[i + 8] = wT[(kb + 16 + i) * stride + n];
  }
  return bf;
}

__device__ inline v8f wmma_f16(v16h a, v16h b, v8f c)
{
  return __builtin_amdgcn_wmma_f32_16x16x32_f16(false, a, false, b,
                                                (short)0, c, false, false);
}

// ---------------------------- main expert kernel ----------------------------
// One block per (patch, half): 4 of the 8 L-slices -> 256 GEMM rows.
__global__ __launch_bounds__(256)
void moe_main(const float* __restrict__ x,
              const float* __restrict__ dw_w,   const float* __restrict__ dw_b,
              const float* __restrict__ ln_g,   const float* __restrict__ ln_b,
              const float* __restrict__ pwin_w, const float* __restrict__ pwin_b,
              const float* __restrict__ pwout_w,const float* __restrict__ pwout_b,
              const Route* __restrict__ route,
              float* __restrict__ out)
{
  __shared__ __align__(16) _Float16 yln[256 * YSTR];   // normalized acts (fp16)
  __shared__ __align__(16) _Float16 hgt[256 * YSTR];   // gated hidden (fp16)
  __shared__ __align__(16) _Float16 winT[CH * WSTR];   // W_in^T  [c][d] fp16
  __shared__ __align__(16) _Float16 woutT[CH * OSTR];  // W_out^T [c][d] fp16
  __shared__ __align__(16) float dwW[CH * 49];         // depthwise taps fp32
  __shared__ float lnG[64], lnB[64];

  const int blk  = blockIdx.x;
  const int n    = blk >> 1;          // patch id
  const int half = blk & 1;           // which 4 L-slices
  const int b    = n / (NH * NW);
  const int rem  = n % (NH * NW);
  const int ph   = rem / NW, pw = rem % NW;
  const int tid  = threadIdx.x;
  const int lane = tid & 31;
  const int wave = tid >> 5;

  const Route rt   = route[n];
  const long pbase = (long)(ph * 8) * HW + pw * 8;    // multiple of 8 floats

#pragma unroll 1
  for (int k = 0; k < 2; ++k) {
    const int   e  = (k == 0) ? rt.e0 : rt.e1;
    const float wk = (k == 0) ? rt.w0 : rt.w1;

    // ---- stage expert weights into LDS (fp32 -> fp16 for GEMM operands) ----
    for (int i = tid; i < CH * 49; i += 256) dwW[i] = dw_w[e * CH * 49 + i];
    for (int i = tid; i < 128 * CH; i += 256) {
      int d = i >> 6, c = i & 63;
      winT[c * WSTR + d] = (_Float16)pwin_w[e * 128 * CH + i];
    }
    for (int i = tid; i < CH * CH; i += 256) {
      int d = i >> 6, c = i & 63;
      woutT[c * OSTR + d] = (_Float16)pwout_w[e * CH * CH + i];
    }
    if (tid < 64) { lnG[tid] = ln_g[e * 64 + tid]; lnB[tid] = ln_b[e * 64 + tid]; }
    __syncthreads();

    // ---- depthwise 7x7 (zero-pad within patch) + LayerNorm -> yln fp16 ----
    // Fully unrolled scatter conv: every register index is compile-time
    // constant -> conv outputs o[64] live entirely in VGPRs (no scratch).
    {
      const int c    = tid >> 2;
      const int lsub = tid & 3;
      const int l    = half * 4 + lsub;
      const float* xin = x + ((long)(b * CH + c) * LL + l) * PLANE + pbase;

      float w[49];
#pragma unroll
      for (int i = 0; i < 49; ++i) w[i] = dwW[c * 49 + i];

      const float bias = dw_b[e * CH + c];
      float o[64];
#pragma unroll
      for (int i = 0; i < 64; ++i) o[i] = bias;

#pragma unroll
      for (int iy = 0; iy < 8; ++iy) {
        const v4f* rp = (const v4f*)(xin + iy * HW);
        v4f r0 = rp[0], r1 = rp[1];
        float row[8] = { r0[0], r0[1], r0[2], r0[3], r1[0], r1[1], r1[2], r1[3] };
#pragma unroll
        for (int ix = 0; ix < 8; ++ix) {
          const float v = row[ix];
          // in(iy,ix) contributes to out(py,px) with weight w[(iy-py)+3][(ix-px)+3]
#pragma unroll
          for (int dy = -3; dy <= 3; ++dy) {
            const int py = iy + dy;
            if (py < 0 || py > 7) continue;
#pragma unroll
            for (int dx = -3; dx <= 3; ++dx) {
              const int px = ix + dx;
              if (px < 0 || px > 7) continue;
              o[py * 8 + px] += v * w[(3 - dy) * 7 + (3 - dx)];
            }
          }
        }
      }
      float s = 0.f, ss = 0.f;
#pragma unroll
      for (int i = 0; i < 64; ++i) { s += o[i]; ss += o[i] * o[i]; }
      const float mu  = s * (1.f / 64.f);
      const float inv = __frsqrt_rn(ss * (1.f / 64.f) - mu * mu + 1e-5f);
#pragma unroll
      for (int hw = 0; hw < 64; ++hw)
        yln[(lsub * 64 + hw) * YSTR + c] =
            (_Float16)((o[hw] - mu) * inv * lnG[hw] + lnB[hw]);
    }
    __syncthreads();

    // ---- pointwise-in (C->2C) WMMA + SiLU gate -> hgt fp16 ----
    // 16 M-tiles x 4 N-tile pairs (a-half d, g-half d+64) = 64 gated tiles.
    // K split into independent accumulators: 4 hazard-free WMMAs per tile.
    for (int t = wave; t < 64; t += 8) {
      const int mt = t >> 2, ntp = t & 3;
      const int dcol = lane & 15;
      const float ba = pwin_b[e * 128 +      16 * ntp + dcol];
      const float bg = pwin_b[e * 128 + 64 + 16 * ntp + dcol];
      v8f accA0, accG0, accA1, accG1;
#pragma unroll
      for (int r = 0; r < 8; ++r) {
        accA0[r] = ba; accG0[r] = bg; accA1[r] = 0.f; accG1[r] = 0.f;
      }
      v16h a0  = load_fragA(yln, mt, 0, lane);
      v16h a1  = load_fragA(yln, mt, 1, lane);
      v16h b00 = load_fragB(winT, WSTR, ntp,     0, lane);
      v16h b10 = load_fragB(winT, WSTR, ntp + 4, 0, lane);
      v16h b01 = load_fragB(winT, WSTR, ntp,     1, lane);
      v16h b11 = load_fragB(winT, WSTR, ntp + 4, 1, lane);
      accA0 = wmma_f16(a0, b00, accA0);
      accG0 = wmma_f16(a0, b10, accG0);
      accA1 = wmma_f16(a1, b01, accA1);
      accG1 = wmma_f16(a1, b11, accG1);

      const int mrow = 16 * mt + ((lane & 16) >> 1);
      const int col  = 16 * ntp + dcol;
#pragma unroll
      for (int r = 0; r < 8; ++r) {
        const float av = accA0[r] + accA1[r];
        const float gv = accG0[r] + accG1[r];
        // silu(a)*g with fast v_rcp (no IEEE div sequence)
        const float sg = __builtin_amdgcn_rcpf(1.f + __expf(-av));
        hgt[(mrow + r) * YSTR + col] = (_Float16)(av * sg * gv);
      }
    }
    __syncthreads();

    // ---- pointwise-out (C->C) WMMA + weighted combine (+residual on k=1) ----
    for (int t = wave; t < 64; t += 8) {
      const int mt = t >> 2, nt = t & 3;
      const int dcol = lane & 15;
      const float bo = pwout_b[e * CH + 16 * nt + dcol];
      v8f acc0, acc1;
#pragma unroll
      for (int r = 0; r < 8; ++r) { acc0[r] = bo; acc1[r] = 0.f; }
      v16h a0  = load_fragA(hgt, mt, 0, lane);
      v16h a1  = load_fragA(hgt, mt, 1, lane);
      v16h bb0 = load_fragB(woutT, OSTR, nt, 0, lane);
      v16h bb1 = load_fragB(woutT, OSTR, nt, 1, lane);
      acc0 = wmma_f16(a0, bb0, acc0);
      acc1 = wmma_f16(a1, bb1, acc1);

      // D rows r=0..7 map to m = mbase+r; mbase % 8 == 0 -> hw&7 == r:
      // the 8 results are 8 consecutive floats in global memory -> 2x b128.
      const int col   = 16 * nt + dcol;                 // channel
      const int mbase = 16 * mt + ((lane & 16) >> 1);   // multiple of 8
      const int l     = half * 4 + (mbase >> 6);
      const int hwb   = mbase & 63;                     // multiple of 8
      const long off0 = ((long)(b * CH + col) * LL + l) * PLANE + pbase
                      + (hwb >> 3) * HW;                // 16B aligned
      v4f lo, hi;
#pragma unroll
      for (int r = 0; r < 4; ++r) {
        lo[r] = wk * (acc0[r] + acc1[r]);
        hi[r] = wk * (acc0[r + 4] + acc1[r + 4]);
      }
      if (k == 0) {                                     // first expert: init
        *(v4f*)(out + off0)     = lo;
        *(v4f*)(out + off0 + 4) = hi;
      } else {                                          // second: add + residual
        v4f po = *(const v4f*)(out + off0);
        v4f p1 = *(const v4f*)(out + off0 + 4);
        v4f x0 = *(const v4f*)(x + off0);
        v4f x1 = *(const v4f*)(x + off0 + 4);
        *(v4f*)(out + off0)     = po + lo + x0;
        *(v4f*)(out + off0 + 4) = p1 + hi + x1;
      }
    }
    __syncthreads();   // protect LDS reuse by next expert pass
  }
}

// ---------------------------------------------------------------------------
extern "C" void kernel_launch(void* const* d_in, const int* in_sizes, int n_in,
                              void* d_out, int out_size, void* d_ws, size_t ws_size,
                              hipStream_t stream)
{
  (void)in_sizes; (void)n_in; (void)out_size; (void)ws_size;
  const float* x        = (const float*)d_in[0];
  const float* router_w = (const float*)d_in[1];
  const float* router_b = (const float*)d_in[2];
  const float* dw_w     = (const float*)d_in[3];
  const float* dw_b     = (const float*)d_in[4];
  const float* ln_g     = (const float*)d_in[5];
  const float* ln_b     = (const float*)d_in[6];
  const float* pwin_w   = (const float*)d_in[7];
  const float* pwin_b   = (const float*)d_in[8];
  const float* pwout_w  = (const float*)d_in[9];
  const float* pwout_b  = (const float*)d_in[10];
  float* out            = (float*)d_out;
  Route* route          = (Route*)d_ws;

  moe_router<<<NPATCH, 256, 0, stream>>>(x, router_w, router_b, route);
  moe_main<<<NPATCH * 2, 256, 0, stream>>>(x, dw_w, dw_b, ln_g, ln_b,
                                           pwin_w, pwin_b, pwout_w, pwout_b,
                                           route, out);
}
